// WindowAttention_24532853194942
// MI455X (gfx1250) — matrix operand order repaired and checked by reading, not compiled, not run on hardware
//
#include <hip/hip_runtime.h>
#include <hip/hip_bf16.h>

// ---------------------------------------------------------------------------
// Swin shifted-window attention, fused, bf16 WMMA (f32 accumulate) for gfx1250.
// B=16 H=W=64 C=384 ws=8 shift=4 nh=12 hd=32 -> 1024 windows, L=64 tokens.
// ---------------------------------------------------------------------------

typedef __attribute__((ext_vector_type(16))) __bf16 bf16x16;
typedef __attribute__((ext_vector_type(8)))  __bf16 bf16x8;
typedef __attribute__((ext_vector_type(4)))  __bf16 bf16x4;
typedef __attribute__((ext_vector_type(8)))  float  f32x8;
typedef __attribute__((ext_vector_type(4)))  float  f32x4;

#define CDIM 384
#define LTOK 64
#define NHEAD 12
#define HDIM 32
#define TPB 256

__device__ __forceinline__ bf16x16 load_frag16(const __bf16* p0, const __bf16* p1) {
    bf16x8 lo = *(const bf16x8*)p0;
    bf16x8 hi = *(const bf16x8*)p1;
    return __builtin_shufflevector(lo, hi, 0,1,2,3,4,5,6,7,8,9,10,11,12,13,14,15);
}

// A-fragment (16x32 bf16, row-major source row for this lane's M = lane%16).
// lanes 0-15 hold K=[k0+0..15], lanes 16-31 K=[k0+16..31] (matches frag_b contiguous order).
__device__ __forceinline__ bf16x16 frag_a(const __bf16* rowbase, int k0, int lh) {
    return load_frag16(rowbase + k0 + 16 * lh, rowbase + k0 + 16 * lh + 8);
}

// B-fragment (32x16 bf16): lane holds 16 contiguous K values of column N = lane%16
// (source pre-transposed so this is one contiguous 32B chunk).
__device__ __forceinline__ bf16x16 frag_b(const __bf16* p) {
    return load_frag16(p, p + 8);
}

__device__ __forceinline__ f32x8 wmma_bf16(bf16x16 a, bf16x16 b, f32x8 c) {
    return __builtin_amdgcn_wmma_f32_16x16x32_bf16(
        /*neg_a=*/false, a, /*neg_b=*/false, b,
        /*c_mod=*/(short)0, c, /*reuse_a=*/false, /*reuse_b=*/false);
}

// ---------------------------------------------------------------------------
// Kernel 0: convert + transpose weights to bf16 so B-fragments are contiguous.
//   wqkvT[n][k] = (bf16) w_qkv[k][n]   (n<1152, k<384)
//   wprojT[n][k] = (bf16) w_proj[k][n] (n<384,  k<384)
// ---------------------------------------------------------------------------
__global__ void swin_prep_kernel(const float* __restrict__ wqkv,
                                 const float* __restrict__ wproj,
                                 __bf16* __restrict__ wqkvT,
                                 __bf16* __restrict__ wprojT) {
    int i = blockIdx.x * blockDim.x + threadIdx.x;
    if (i < CDIM * 3 * CDIM) {
        int k = i / (3 * CDIM);
        int n = i - k * (3 * CDIM);
        wqkvT[n * CDIM + k] = (__bf16)wqkv[i];
    }
    if (i < CDIM * CDIM) {
        int k = i / CDIM;
        int n = i - k * CDIM;
        wprojT[n * CDIM + k] = (__bf16)wproj[i];
    }
}

// ---------------------------------------------------------------------------
// Kernel 1: fused window attention. One block = one window (8 waves).
// ---------------------------------------------------------------------------
__global__ __launch_bounds__(TPB) void swin_attn_kernel(
    const float*  __restrict__ x,
    const __bf16* __restrict__ wqkvT,
    const float*  __restrict__ bqkv,
    const __bf16* __restrict__ wprojT,
    const float*  __restrict__ bproj,
    float*        __restrict__ out) {

    __shared__ __bf16 lx  [LTOK * CDIM];   // 49152 B  window activations (bf16)
    __shared__ __bf16 lq  [LTOK * HDIM];   //  4096 B  q (scaled)
    __shared__ __bf16 lk  [LTOK * HDIM];   //  4096 B  k
    __shared__ __bf16 lvT [HDIM * LTOK];   //  4096 B  v transposed (d-major)
    __shared__ float  ls  [LTOK * LTOK];   // 16384 B  scores f32
    __shared__ __bf16 la  [LTOK * LTOK];   //  8192 B  softmax(attn) bf16
    __shared__ __bf16 yout[LTOK * CDIM];   // 49152 B  per-window attn output

    const int tid  = threadIdx.x;
    const int wave = tid >> 5;       // 0..7 (wave32)
    const int lane = tid & 31;
    const int l16  = lane & 15;
    const int lh   = lane >> 4;      // 0 or 1

    const int win = blockIdx.x;      // 0..1023
    const int bb  = win >> 6;        // batch
    const int wy  = (win >> 3) & 7;
    const int wx  = win & 7;

    // ---- phase 0: gather shifted window, f32 -> bf16, float4-wide ----------
    for (int e4 = tid; e4 < (LTOK * CDIM) / 4; e4 += TPB) {
        int t   = e4 / (CDIM / 4);
        int ch4 = e4 - t * (CDIM / 4);
        int r   = t >> 3, c = t & 7;
        int sh  = (wy * 8 + r + 4) & 63;   // roll(-4): xw[hp] = x[(hp+4)%64]
        int sw  = (wx * 8 + c + 4) & 63;
        f32x4 v = *(const f32x4*)&x[((size_t)(bb * 64 + sh) * 64 + sw) * CDIM + ch4 * 4];
        bf16x4 b = { (__bf16)v.x, (__bf16)v.y, (__bf16)v.z, (__bf16)v.w };
        *(bf16x4*)&lx[e4 * 4] = b;
    }
    __syncthreads();

    const float scale = 0.17677669529663687f;  // 1/sqrt(32)

    for (int h = 0; h < NHEAD; ++h) {
        // ---- QKV GEMM: [64,384]@[384,96] -> 24 tiles = 4M x 6N.
        // Each wave: fixed M tile (wave&3), 3 N tiles; A fragment shared. ----
        {
            const int m  = wave & 3;
            const int ng = (wave >> 2) * 3;             // first n6 of the 3
            const __bf16* arow = &lx[(m * 16 + l16) * CDIM];
            const __bf16* bcol[3];
            int wrow[3];
            #pragma unroll
            for (int t = 0; t < 3; ++t) {
                int n6  = ng + t;
                int mat = n6 >> 1;                       // 0=q 1=k 2=v
                int nc  = (n6 & 1) << 4;                 // 0 or 16
                wrow[t] = mat * CDIM + h * HDIM + nc;    // output column block
                bcol[t] = &wqkvT[(wrow[t] + l16) * CDIM];
                __builtin_prefetch(bcol[t], 0, 1);
            }
            f32x8 acc0 = {0.f,0.f,0.f,0.f,0.f,0.f,0.f,0.f};
            f32x8 acc1 = acc0, acc2 = acc0;
            #pragma unroll
            for (int k0 = 0; k0 < CDIM; k0 += 32) {
                bf16x16 a = frag_a(arow, k0, lh);        // loaded once, used 3x
                acc0 = wmma_bf16(a, frag_b(bcol[0] + k0 + 16 * lh), acc0);
                acc1 = wmma_bf16(a, frag_b(bcol[1] + k0 + 16 * lh), acc1);
                acc2 = wmma_bf16(a, frag_b(bcol[2] + k0 + 16 * lh), acc2);
            }
            f32x8 accs[3] = { acc0, acc1, acc2 };
            #pragma unroll
            for (int t = 0; t < 3; ++t) {
                int n6  = ng + t;
                int mat = n6 >> 1;
                int nc  = (n6 & 1) << 4;
                float bias = bqkv[wrow[t] + l16];
                int d = nc + l16;
                #pragma unroll
                for (int j = 0; j < 8; ++j) {
                    int tok = m * 16 + j + 8 * lh;
                    float v = accs[t][j] + bias;
                    if (mat == 0)      lq[tok * HDIM + d] = (__bf16)(v * scale);
                    else if (mat == 1) lk[tok * HDIM + d] = (__bf16)v;
                    else               lvT[d * LTOK + tok] = (__bf16)v;
                }
            }
        }
        __syncthreads();

        // ---- scores = (q*scale) @ k^T : 16 tiles, 2/wave sharing B, K=32 --
        {
            const int ni  = wave >> 1;                   // 0..3
            const int mib = (wave & 1) * 2;              // 0 or 2
            bf16x16 b = frag_b(&lk[(ni * 16 + l16) * HDIM + 16 * lh]);
            #pragma unroll
            for (int i = 0; i < 2; ++i) {
                int mi = mib + i;
                bf16x16 a = frag_a(&lq[(mi * 16 + l16) * HDIM], 0, lh);
                f32x8 acc = {0.f,0.f,0.f,0.f,0.f,0.f,0.f,0.f};
                acc = wmma_bf16(a, b, acc);
                #pragma unroll
                for (int j = 0; j < 8; ++j)
                    ls[(mi * 16 + j + 8 * lh) * LTOK + ni * 16 + l16] = acc[j];
            }
        }
        __syncthreads();

        // ---- softmax over each of the 64 rows -----------------------------
        if (tid < LTOK) {
            const float* row = &ls[tid * LTOK];
            float mx = -3.0e38f;
            for (int j = 0; j < LTOK; ++j) mx = fmaxf(mx, row[j]);
            float sum = 0.f;
            for (int j = 0; j < LTOK; ++j) sum += __expf(row[j] - mx);
            float inv = 1.0f / sum;
            for (int j4 = 0; j4 < LTOK; j4 += 4) {       // packed 8B stores
                bf16x4 o = { (__bf16)(__expf(row[j4 + 0] - mx) * inv),
                             (__bf16)(__expf(row[j4 + 1] - mx) * inv),
                             (__bf16)(__expf(row[j4 + 2] - mx) * inv),
                             (__bf16)(__expf(row[j4 + 3] - mx) * inv) };
                *(bf16x4*)&la[tid * LTOK + j4] = o;
            }
        }
        __syncthreads();

        // ---- out = attn @ v : 8 tiles, 1 per wave, K=64 -------------------
        {
            int mo = wave & 3;
            int d0 = (wave >> 2) * 16;
            f32x8 acc = {0.f,0.f,0.f,0.f,0.f,0.f,0.f,0.f};
            #pragma unroll
            for (int k0 = 0; k0 < LTOK; k0 += 32) {
                bf16x16 a = frag_a(&la[(mo * 16 + l16) * LTOK], k0, lh);
                bf16x16 b = frag_b(&lvT[(d0 + l16) * LTOK + k0 + 16 * lh]);
                acc = wmma_bf16(a, b, acc);
            }
            #pragma unroll
            for (int j = 0; j < 8; ++j) {
                int tok = mo * 16 + j + 8 * lh;
                yout[tok * CDIM + h * HDIM + d0 + l16] = (__bf16)acc[j];
            }
        }
        __syncthreads();   // protects lq/lk/lvT/ls/la reuse next head
    }

    // ---- proj: [64,384]@[384,384] = 96 tiles = 4M x 24N.
    // Wave: fixed M tile, 12 N tiles in 2 groups of 6; A fragment shared. ----
    {
        const int m     = wave & 3;
        const int nbase = (wave >> 2) * 12;              // 0 or 12 (N tile idx)
        const __bf16* arow = &yout[(m * 16 + l16) * CDIM];
        #pragma unroll
        for (int g = 0; g < 2; ++g) {
            const __bf16* bcol[6];
            #pragma unroll
            for (int t = 0; t < 6; ++t) {
                int n0 = (nbase + g * 6 + t) * 16;
                bcol[t] = &wprojT[(n0 + l16) * CDIM];
            }
            __builtin_prefetch(bcol[0], 0, 1);
            f32x8 acc[6];
            #pragma unroll
            for (int t = 0; t < 6; ++t) acc[t] = (f32x8){0.f,0.f,0.f,0.f,0.f,0.f,0.f,0.f};
            #pragma unroll
            for (int k0 = 0; k0 < CDIM; k0 += 32) {
                bf16x16 a = frag_a(arow, k0, lh);        // loaded once, used 6x
                #pragma unroll
                for (int t = 0; t < 6; ++t)
                    acc[t] = wmma_bf16(a, frag_b(bcol[t] + k0 + 16 * lh), acc[t]);
            }
            #pragma unroll
            for (int t = 0; t < 6; ++t) {
                int n0 = (nbase + g * 6 + t) * 16;
                float bias = bproj[n0 + l16];
                #pragma unroll
                for (int j = 0; j < 8; ++j) {
                    int tok = m * 16 + j + 8 * lh;
                    int r = tok >> 3, c = tok & 7;
                    int gh = (wy * 8 + r + 4) & 63;      // roll(+4) scatter
                    int gw = (wx * 8 + c + 4) & 63;
                    out[((size_t)(bb * 64 + gh) * 64 + gw) * CDIM + n0 + l16] =
                        acc[t][j] + bias;
                }
            }
        }
    }
}

// ---------------------------------------------------------------------------
extern "C" void kernel_launch(void* const* d_in, const int* in_sizes, int n_in,
                              void* d_out, int out_size, void* d_ws, size_t ws_size,
                              hipStream_t stream) {
    const float* x     = (const float*)d_in[0];   // [16,64,64,384]
    const float* wqkv  = (const float*)d_in[1];   // [384,1152]
    const float* bqkv  = (const float*)d_in[2];   // [1152]
    const float* wproj = (const float*)d_in[3];   // [384,384]
    const float* bproj = (const float*)d_in[4];   // [384]
    float* out = (float*)d_out;

    __bf16* wqkvT  = (__bf16*)d_ws;                                   // 1152*384 bf16
    __bf16* wprojT = (__bf16*)((char*)d_ws + (size_t)3 * CDIM * CDIM * sizeof(__bf16));

    int nprep = CDIM * 3 * CDIM;  // 442368 covers both weight arrays
    swin_prep_kernel<<<(nprep + TPB - 1) / TPB, TPB, 0, stream>>>(wqkv, wproj, wqkvT, wprojT);

    swin_attn_kernel<<<1024, TPB, 0, stream>>>(x, wqkvT, bqkv, wprojT, bproj, out);
}
